// GatedAttentionUnit_81312320848005
// MI455X (gfx1250) — compile-verified, matrix operand order checked by reading, not run on hardware
//
#include <hip/hip_runtime.h>
#include <hip/hip_bf16.h>
#include <cmath>

typedef __attribute__((ext_vector_type(16))) _Float16 v16h;
typedef __attribute__((ext_vector_type(8)))  _Float16 v8h;
typedef __attribute__((ext_vector_type(8)))  float    v8f;

#define BDIM 4
#define NSEQ 4096
#define HDIM 1024
#define IDIM 2048
#define DKD  128
#define BN   (BDIM * NSEQ)      // 16384 rows total
#define FDIM (2 * IDIM + DKD)   // 4224

// ---------------------------------------------------------------------------
// Fragment loader for CDNA5 16-bit 16x32 A-layout (and the matching 32x16
// B-layout when the staging buffer is stored [col][k]):
//   lane = 16*h + r  holds row/col r, K = h*8..h*8+7 (lo) and 16+h*8.. (hi)
// ---------------------------------------------------------------------------
__device__ __forceinline__ v16h frag16(const _Float16* p, int khalf) {
  v8h lo = *(const v8h*)(p + khalf * 8);
  v8h hi = *(const v8h*)(p + khalf * 8 + 16);
  v16h f;
#pragma unroll
  for (int i = 0; i < 8; ++i) { f[i] = lo[i]; f[i + 8] = hi[i]; }
  return f;
}

__device__ __forceinline__ v8f wmma_f16(v16h a, v16h b, v8f c) {
  return __builtin_amdgcn_wmma_f32_16x16x32_f16(false, a, false, b,
                                                (short)0, c, false, false);
}

__device__ __forceinline__ float fast_rcp(float x) {
  return __builtin_amdgcn_rcpf(x);      // v_rcp_f32
}

__device__ __forceinline__ float silu(float x) {
  return x * fast_rcp(1.f + __expf(-x));
}

// Pack two floats into adjacent f16s and store as one 32-bit LDS write.
__device__ __forceinline__ void store_pk16(_Float16* dst, float a, float b) {
  union { _Float16 h[2]; unsigned int u; } pk;
  pk.h[0] = (_Float16)a;
  pk.h[1] = (_Float16)b;
  *(unsigned int*)dst = pk.u;           // v_cvt_pk_f16_f32 + ds_store_b32
}

// ---------------------------------------------------------------------------
// K0: per-batch softmax temperature: log(sum(mask)) / log(512) / sqrt(128)
// ---------------------------------------------------------------------------
__global__ void mask_scale_kernel(const int* __restrict__ mask,
                                  float* __restrict__ scale) {
  __shared__ int red[256];
  int b = blockIdx.x, t = threadIdx.x;
  int s = 0;
  for (int n = t; n < NSEQ; n += 256) s += mask[b * NSEQ + n];
  red[t] = s;
  __syncthreads();
  for (int o = 128; o > 0; o >>= 1) {
    if (t < o) red[t] += red[t + o];
    __syncthreads();
  }
  if (t == 0)
    scale[b] = (logf((float)red[0]) / 6.2383246250395068f) * 0.0883883476483184f;
}

// ---------------------------------------------------------------------------
// K1: X = silu(HS @ Wi); split into U (f16 [BN][I]), Vt (f16 [I][BN]), Z (f32)
//     128x128 tile per WG, 8 waves in 2x4, f16 WMMA with f32 accum.
// ---------------------------------------------------------------------------
__global__ __launch_bounds__(256, 1) void gemm1_silu_kernel(
    const float* __restrict__ HS, const float* __restrict__ Wi,
    _Float16* __restrict__ U, _Float16* __restrict__ Vt, float* __restrict__ Z) {
  __shared__ __align__(16) _Float16 As[128][32];  // [m][k]
  __shared__ __align__(16) _Float16 Bs[128][32];  // [n][k]  (transposed Wi)
  int m0 = blockIdx.y * 128, n0 = blockIdx.x * 128;
  int t = threadIdx.x, lane = t & 31, wid = t >> 5;
  int wm = wid & 1, wn = wid >> 1;          // 2 x 4 wave grid
  int ln = lane & 15, lh = lane >> 4;

  v8f acc[4][2];
#pragma unroll
  for (int a = 0; a < 4; ++a)
#pragma unroll
    for (int bq = 0; bq < 2; ++bq)
#pragma unroll
      for (int e = 0; e < 8; ++e) acc[a][bq][e] = 0.f;

  for (int k0 = 0; k0 < HDIM; k0 += 32) {
    {  // stage A: 128x32 fp32 -> f16 (vectorized converts + b128 DS stores)
      int m = t >> 1, half = t & 1;
      const float* src = HS + (size_t)(m0 + m) * HDIM + k0 + half * 16;
      if (k0 + 32 < HDIM) __builtin_prefetch(src + 32, 0, 1);
      v8h h0, h1;
#pragma unroll
      for (int j = 0; j < 8; ++j) { h0[j] = (_Float16)src[j]; h1[j] = (_Float16)src[j + 8]; }
      *(v8h*)&As[m][half * 16] = h0;
      *(v8h*)&As[m][half * 16 + 8] = h1;
    }
    {  // stage B transposed via k-pairs: Bs[n][2kp..2kp+1] as one b32 store
      int kp = t >> 4, nc = t & 15;           // kp: 0..15 (k-pair), nc: 0..15
      const float* s0 = Wi + (size_t)(k0 + 2 * kp) * FDIM + n0 + nc * 8;
      const float* s1 = s0 + FDIM;
      if (k0 + 32 < HDIM) __builtin_prefetch(s0 + 32 * FDIM, 0, 1);
#pragma unroll
      for (int j = 0; j < 8; ++j)
        store_pk16(&Bs[nc * 8 + j][2 * kp], s0[j], s1[j]);
    }
    __syncthreads();
    v16h af[4], bf[2];
#pragma unroll
    for (int mt = 0; mt < 4; ++mt) af[mt] = frag16(&As[wm * 64 + mt * 16 + ln][0], lh);
#pragma unroll
    for (int nt = 0; nt < 2; ++nt) bf[nt] = frag16(&Bs[wn * 32 + nt * 16 + ln][0], lh);
#pragma unroll
    for (int mt = 0; mt < 4; ++mt)
#pragma unroll
      for (int nt = 0; nt < 2; ++nt) acc[mt][nt] = wmma_f16(af[mt], bf[nt], acc[mt][nt]);
    __syncthreads();
  }

  // Epilogue: whole block lands in exactly one region (uniform branch).
  if (n0 < IDIM) {
#pragma unroll
    for (int mt = 0; mt < 4; ++mt)
#pragma unroll
      for (int nt = 0; nt < 2; ++nt)
#pragma unroll
        for (int e = 0; e < 8; ++e) {
          int row = m0 + wm * 64 + mt * 16 + e + 8 * lh;
          int col = n0 + wn * 32 + nt * 16 + ln;
          U[(size_t)row * IDIM + col] = (_Float16)silu(acc[mt][nt][e]);
        }
  } else if (n0 < 2 * IDIM) {
#pragma unroll
    for (int mt = 0; mt < 4; ++mt)
#pragma unroll
      for (int nt = 0; nt < 2; ++nt)
#pragma unroll
        for (int e = 0; e < 8; ++e) {
          int row = m0 + wm * 64 + mt * 16 + e + 8 * lh;
          int col = n0 - IDIM + wn * 32 + nt * 16 + ln;
          Vt[(size_t)col * BN + row] = (_Float16)silu(acc[mt][nt][e]);
        }
  } else {
#pragma unroll
    for (int mt = 0; mt < 4; ++mt)
#pragma unroll
      for (int nt = 0; nt < 2; ++nt)
#pragma unroll
        for (int e = 0; e < 8; ++e) {
          int row = m0 + wm * 64 + mt * 16 + e + 8 * lh;
          int col = n0 - 2 * IDIM + wn * 32 + nt * 16 + ln;
          Z[(size_t)row * DKD + col] = silu(acc[mt][nt][e]);
        }
  }
}

// ---------------------------------------------------------------------------
// K2: rotary embedding for q (x_gcn) and k (Z * k_scale), f16 out
// ---------------------------------------------------------------------------
__global__ void rotary_kernel(const float* __restrict__ x_gcn,
                              const float* __restrict__ Z,
                              const float* __restrict__ sinb,
                              const float* __restrict__ cosb,
                              const float* __restrict__ kscale,
                              _Float16* __restrict__ qr, _Float16* __restrict__ kr) {
  int bn = blockIdx.x, d = threadIdx.x;              // d in [0,64)
  int n = bn & (NSEQ - 1);
  float s = sinb[n * 64 + d], c = cosb[n * 64 + d];
  const float* q = x_gcn + (size_t)bn * DKD;
  float q1 = q[2 * d], q2 = q[2 * d + 1];
  qr[(size_t)bn * DKD + d]      = (_Float16)(q1 * c - q2 * s);
  qr[(size_t)bn * DKD + 64 + d] = (_Float16)(q2 * c + q1 * s);
  const float* z = Z + (size_t)bn * DKD;
  float k1 = z[2 * d] * kscale[2 * d], k2 = z[2 * d + 1] * kscale[2 * d + 1];
  kr[(size_t)bn * DKD + d]      = (_Float16)(k1 * c - k2 * s);
  kr[(size_t)bn * DKD + 64 + d] = (_Float16)(k2 * c + k1 * s);
}

// ---------------------------------------------------------------------------
// K3: softmax row stats via streamed QK^T WMMA.
//     Per-lane private online (max,sum) over this lane's columns only; one
//     log2(16)-step (m,s) merge at the end -> zero inner-loop shuffles.
// ---------------------------------------------------------------------------
__global__ __launch_bounds__(256, 1) void attn_stats_kernel(
    const _Float16* __restrict__ qr, const _Float16* __restrict__ kr,
    const int* __restrict__ mask, const float* __restrict__ scale,
    float* __restrict__ rowmax, float* __restrict__ rowsum) {
  int t = threadIdx.x, lane = t & 31, wid = t >> 5;
  int m0 = blockIdx.x * 128 + wid * 16;
  int b = m0 >> 12;
  int ln = lane & 15, lh = lane >> 4;
  float scl = scale[b];

  v16h qf[4];
#pragma unroll
  for (int kk = 0; kk < 4; ++kk)
    qf[kk] = frag16(qr + (size_t)(m0 + ln) * DKD + kk * 32, lh);

  float rmax[8], rsum[8];
#pragma unroll
  for (int e = 0; e < 8; ++e) { rmax[e] = -3.0e38f; rsum[e] = 0.f; }

  const _Float16* kbase = kr + (size_t)b * NSEQ * DKD;
  const int* mbase = mask + b * NSEQ;

  for (int nt = 0; nt < NSEQ / 16; ++nt) {
    v8f sacc;
#pragma unroll
    for (int e = 0; e < 8; ++e) sacc[e] = 0.f;
#pragma unroll
    for (int kk = 0; kk < 4; ++kk) {
      v16h kf = frag16(kbase + (size_t)(nt * 16 + ln) * DKD + kk * 32, lh);
      sacc = wmma_f16(qf[kk], kf, sacc);
    }
    int mk = mbase[nt * 16 + ln];
#pragma unroll
    for (int e = 0; e < 8; ++e) {
      float sv = sacc[e] * scl;
      float nm = mk ? fmaxf(rmax[e], sv) : rmax[e];
      float pe = mk ? __expf(sv - nm) : 0.f;       // exp(-inf)->0 handles init
      rsum[e] = rsum[e] * __expf(rmax[e] - nm) + pe;
      rmax[e] = nm;
    }
  }
  // merge (max,sum) pairs across the 16 lanes of each half-wave
#pragma unroll
  for (int o = 1; o < 16; o <<= 1) {
#pragma unroll
    for (int e = 0; e < 8; ++e) {
      float om = __shfl_xor(rmax[e], o, 16);
      float os = __shfl_xor(rsum[e], o, 16);
      float nm = fmaxf(rmax[e], om);
      rsum[e] = rsum[e] * __expf(rmax[e] - nm) + os * __expf(om - nm);
      rmax[e] = nm;
    }
  }
  if (ln == 0) {
#pragma unroll
    for (int e = 0; e < 8; ++e) {
      rowmax[m0 + 8 * lh + e] = rmax[e];
      rowsum[m0 + 8 * lh + e] = rsum[e];
    }
  }
}

// ---------------------------------------------------------------------------
// K4: G = u * (softmax(S) @ V).  WG = 16 rows x 1024 I-cols (grid.y halves).
//     Per 128-key chunk: each wave builds one 16x16 P tile -> LDS [m][n],
//     then all waves run P @ V^T (Vt gives contiguous-K B-fragments).
// ---------------------------------------------------------------------------
__global__ __launch_bounds__(256, 1) void attn_pv_kernel(
    const _Float16* __restrict__ qr, const _Float16* __restrict__ kr,
    const _Float16* __restrict__ Vt, const _Float16* __restrict__ U,
    const int* __restrict__ mask, const float* __restrict__ scale,
    const float* __restrict__ rowmax, const float* __restrict__ rowsum,
    _Float16* __restrict__ G) {
  __shared__ __align__(16) _Float16 P[16][128];
  int t = threadIdx.x, lane = t & 31, wid = t >> 5;
  int m0 = blockIdx.x * 16;
  int b = m0 >> 12;
  int ibase = blockIdx.y * 1024 + wid * 128;
  int ln = lane & 15, lh = lane >> 4;
  float scl = scale[b];

  v16h qf[4];
#pragma unroll
  for (int kk = 0; kk < 4; ++kk)
    qf[kk] = frag16(qr + (size_t)(m0 + ln) * DKD + kk * 32, lh);

  float rm[8];
#pragma unroll
  for (int e = 0; e < 8; ++e) rm[e] = rowmax[m0 + 8 * lh + e];

  v8f acc[8];
#pragma unroll
  for (int it = 0; it < 8; ++it)
#pragma unroll
    for (int e = 0; e < 8; ++e) acc[it][e] = 0.f;

  const _Float16* kbase = kr + (size_t)b * NSEQ * DKD;
  const int* mbase = mask + b * NSEQ;
  const _Float16* vtb = Vt + (size_t)b * NSEQ;

  for (int c0 = 0; c0 < NSEQ; c0 += 128) {
    // phase 1: this wave's 16-key P tile
    int ns = c0 + wid * 16;
    v8f sacc;
#pragma unroll
    for (int e = 0; e < 8; ++e) sacc[e] = 0.f;
#pragma unroll
    for (int kk = 0; kk < 4; ++kk) {
      v16h kf = frag16(kbase + (size_t)(ns + ln) * DKD + kk * 32, lh);
      sacc = wmma_f16(qf[kk], kf, sacc);
    }
    int mk = mbase[ns + ln];
#pragma unroll
    for (int e = 0; e < 8; ++e) {
      float pv = mk ? __expf(sacc[e] * scl - rm[e]) : 0.f;
      P[e + 8 * lh][wid * 16 + ln] = (_Float16)pv;
    }
    __syncthreads();
    // phase 2: P(16x128) @ V^T-slice
    v16h a2[4];
#pragma unroll
    for (int kk = 0; kk < 4; ++kk) a2[kk] = frag16(&P[ln][kk * 32], lh);
#pragma unroll
    for (int it = 0; it < 8; ++it) {
#pragma unroll
      for (int kk = 0; kk < 4; ++kk) {
        v16h b2 = frag16(vtb + (size_t)(ibase + it * 16 + ln) * BN + c0 + kk * 32, lh);
        acc[it] = wmma_f16(a2[kk], b2, acc[it]);
      }
    }
    __syncthreads();
  }

  float rinv[8];
#pragma unroll
  for (int e = 0; e < 8; ++e) rinv[e] = fast_rcp(rowsum[m0 + 8 * lh + e]);
#pragma unroll
  for (int it = 0; it < 8; ++it)
#pragma unroll
    for (int e = 0; e < 8; ++e) {
      int row = m0 + e + 8 * lh;
      int col = ibase + it * 16 + ln;
      float g = acc[it][e] * rinv[e] * (float)U[(size_t)row * IDIM + col];
      G[(size_t)row * IDIM + col] = (_Float16)g;
    }
}

// ---------------------------------------------------------------------------
// K5: out = G @ Wo  (f16 A, Wo staged f32->f16 transposed, fp32 out)
// ---------------------------------------------------------------------------
__global__ __launch_bounds__(256, 1) void gemm2_kernel(
    const _Float16* __restrict__ G, const float* __restrict__ Wo,
    float* __restrict__ out) {
  __shared__ __align__(16) _Float16 As[128][32];
  __shared__ __align__(16) _Float16 Bs[128][32];
  int m0 = blockIdx.y * 128, n0 = blockIdx.x * 128;
  int t = threadIdx.x, lane = t & 31, wid = t >> 5;
  int wm = wid & 1, wn = wid >> 1;
  int ln = lane & 15, lh = lane >> 4;

  v8f acc[4][2];
#pragma unroll
  for (int a = 0; a < 4; ++a)
#pragma unroll
    for (int bq = 0; bq < 2; ++bq)
#pragma unroll
      for (int e = 0; e < 8; ++e) acc[a][bq][e] = 0.f;

  for (int k0 = 0; k0 < IDIM; k0 += 32) {
    {  // stage A (already f16)
      int m = t >> 1, half = t & 1;
      const _Float16* src = G + (size_t)(m0 + m) * IDIM + k0 + half * 16;
      if (k0 + 32 < IDIM) __builtin_prefetch(src + 32, 0, 1);
      *(v8h*)&As[m][half * 16]     = *(const v8h*)src;
      *(v8h*)&As[m][half * 16 + 8] = *(const v8h*)(src + 8);
    }
    {  // stage B transposed via k-pairs
      int kp = t >> 4, nc = t & 15;
      const float* s0 = Wo + (size_t)(k0 + 2 * kp) * HDIM + n0 + nc * 8;
      const float* s1 = s0 + HDIM;
      if (k0 + 32 < IDIM) __builtin_prefetch(s0 + 32 * HDIM, 0, 1);
#pragma unroll
      for (int j = 0; j < 8; ++j)
        store_pk16(&Bs[nc * 8 + j][2 * kp], s0[j], s1[j]);
    }
    __syncthreads();
    v16h af[4], bf[2];
#pragma unroll
    for (int mt = 0; mt < 4; ++mt) af[mt] = frag16(&As[wm * 64 + mt * 16 + ln][0], lh);
#pragma unroll
    for (int nt = 0; nt < 2; ++nt) bf[nt] = frag16(&Bs[wn * 32 + nt * 16 + ln][0], lh);
#pragma unroll
    for (int mt = 0; mt < 4; ++mt)
#pragma unroll
      for (int nt = 0; nt < 2; ++nt) acc[mt][nt] = wmma_f16(af[mt], bf[nt], acc[mt][nt]);
    __syncthreads();
  }

#pragma unroll
  for (int mt = 0; mt < 4; ++mt)
#pragma unroll
    for (int nt = 0; nt < 2; ++nt)
#pragma unroll
      for (int e = 0; e < 8; ++e) {
        int row = m0 + wm * 64 + mt * 16 + e + 8 * lh;
        int col = n0 + wn * 32 + nt * 16 + ln;
        out[(size_t)row * HDIM + col] = acc[mt][nt][e];
      }
}

// ---------------------------------------------------------------------------
extern "C" void kernel_launch(void* const* d_in, const int* in_sizes, int n_in,
                              void* d_out, int out_size, void* d_ws, size_t ws_size,
                              hipStream_t stream) {
  const float* HS     = (const float*)d_in[0];
  const float* x_gcn  = (const float*)d_in[1];
  const int*   mask   = (const int*)d_in[2];
  const float* sinb   = (const float*)d_in[3];
  const float* cosb   = (const float*)d_in[4];
  const float* Wi     = (const float*)d_in[5];
  const float* Wo     = (const float*)d_in[6];
  const float* kscale = (const float*)d_in[7];
  float* out = (float*)d_out;

  char* ws = (char*)d_ws;
  size_t off = 0;
  auto alloc = [&](size_t bytes) {
    char* p = ws + off;
    off += (bytes + 255) & ~(size_t)255;
    return p;
  };
  _Float16* U      = (_Float16*)alloc((size_t)BN * IDIM * sizeof(_Float16));
  _Float16* Vt     = (_Float16*)alloc((size_t)IDIM * BN * sizeof(_Float16));
  _Float16* G      = (_Float16*)alloc((size_t)BN * IDIM * sizeof(_Float16));
  float*    Z      = (float*)alloc((size_t)BN * DKD * sizeof(float));
  _Float16* qr     = (_Float16*)alloc((size_t)BN * DKD * sizeof(_Float16));
  _Float16* kr     = (_Float16*)alloc((size_t)BN * DKD * sizeof(_Float16));
  float*    rowmax = (float*)alloc((size_t)BN * sizeof(float));
  float*    rowsum = (float*)alloc((size_t)BN * sizeof(float));
  float*    scl    = (float*)alloc(256);

  mask_scale_kernel<<<BDIM, 256, 0, stream>>>(mask, scl);
  gemm1_silu_kernel<<<dim3(FDIM / 128, BN / 128), 256, 0, stream>>>(HS, Wi, U, Vt, Z);
  rotary_kernel<<<BN, 64, 0, stream>>>(x_gcn, Z, sinb, cosb, kscale, qr, kr);
  attn_stats_kernel<<<BN / 128, 256, 0, stream>>>(qr, kr, mask, scl, rowmax, rowsum);
  attn_pv_kernel<<<dim3(BN / 16, 2), 256, 0, stream>>>(qr, kr, Vt, U, mask, scl,
                                                       rowmax, rowsum, G);
  gemm2_kernel<<<dim3(HDIM / 128, BN / 128), 256, 0, stream>>>(G, Wo, out);
}
